// HeteroGATEncoder1Conv2LinearDropout_15805479649920
// MI455X (gfx1250) — matrix operand order repaired and checked by reading, not compiled, not run on hardware
//
#include <hip/hip_runtime.h>
#include <hip/hip_bf16.h>
#include <math.h>

#define N_NODES 50000
#define DIN     128
#define NH      8
#define NC      32
#define HC      256
#define NE      400000
#define OUT_CH  64
#define SLOPE   0.2f

typedef __attribute__((ext_vector_type(16))) _Float16 v16h;
typedef __attribute__((ext_vector_type(8)))  _Float16 v8h;
typedef __attribute__((ext_vector_type(2)))  _Float16 v2h;
typedef __attribute__((ext_vector_type(8)))  float    v8f;

// ---------------------------------------------------------------- utilities
__global__ void fill_f32(float* __restrict__ p, float v, int n) {
    int i = blockIdx.x * blockDim.x + threadIdx.x;
    if (i < n) p[i] = v;
}

__global__ void cvt_f32_to_f16(const float* __restrict__ src, _Float16* __restrict__ dst, int n) {
    int i = blockIdx.x * blockDim.x + threadIdx.x;
    if (i < n) dst[i] = (_Float16)src[i];
}

__device__ __forceinline__ float atomicMaxFloat(float* addr, float value) {
    // two-sided int trick -> global_atomic_max_i32 / global_atomic_min_u32
    if (value >= 0.0f)
        return __int_as_float(atomicMax((int*)addr, __float_as_int(value)));
    else
        return __uint_as_float(atomicMin((unsigned int*)addr, __float_as_uint(value)));
}

// ---------------------------------------------------------------- B pre-pack
// Pack f32 B[K,Ncols] into WMMA B-fragment order (f16):
//   Bp[(((nt*KT + kt)*32 + lane)*16) + e] = B[(kt*32 + 16*(e>>3) + 8*(lane>>4) + (e&7))*Ncols + nt*16 + (lane&15)]
// so each lane's 16 halves for one wmma are contiguous (2x b128 loads).
__global__ void pack_B_kernel(const float* __restrict__ B, _Float16* __restrict__ Bp,
                              int K, int Ncols) {
    const int t = blockIdx.x * blockDim.x + threadIdx.x;
    if (t >= K * Ncols) return;
    const int KT   = K >> 5;
    const int e    = t & 15;
    const int lane = (t >> 4) & 31;
    const int ktnt = t >> 9;                 // (nt*KT + kt)
    const int kt   = ktnt % KT;
    const int nt   = ktnt / KT;
    const int col  = nt * 16 + (lane & 15);
    const int kb   = kt * 32 + ((e >> 3) << 4) + ((lane >> 4) & 1) * 8 + (e & 7);
    Bp[t] = (_Float16)B[(size_t)kb * Ncols + col];
}

// ---------------------------------------------------------------- WMMA GEMM
// One wave32 per 16-row M-tile; A fragments loaded once (b128), then loop over
// all N-tiles with pre-packed B (L2-resident). KT = K/32, NT = Ncols/16.
// act: 0 = identity, 1 = ELU. Requires M % 16 == 0.
template <int KT, int NT>
__global__ void wmma_gemm_f16(const _Float16* __restrict__ A,    // [M, KT*32] row-major f16
                              const _Float16* __restrict__ Bp,   // packed fragments
                              const float*    __restrict__ bias, // [NT*16]
                              float*          __restrict__ outF32,
                              _Float16*       __restrict__ outF16,
                              int Mtiles, int act) {
    const int lane = threadIdx.x & 31;
    const int tm   = blockIdx.x * blockDim.y + threadIdx.y;     // wave-uniform
    if (tm >= Mtiles) return;                                   // whole-wave exit, EXEC stays ~0
    const int hi   = (lane >> 4) & 1;
    const int row  = tm * 16 + (lane & 15);
    constexpr int K     = KT * 32;
    constexpr int Ncols = NT * 16;

    // A fragments: lane holds row (lane&15); halves {K0..K0+7} and {K0+16..K0+23},
    // K0 = kt*32 + 8*hi -> two aligned 16B chunks per k-step.
    v16h afrag[KT];
    const v8h* __restrict__ Ar = (const v8h*)(A + (size_t)row * K);
    #pragma unroll
    for (int kt = 0; kt < KT; ++kt) {
        v8h lo = Ar[kt * 4 + hi];
        v8h hh = Ar[kt * 4 + 2 + hi];
        afrag[kt] = __builtin_shufflevector(lo, hh, 0, 1, 2, 3, 4, 5, 6, 7,
                                            8, 9, 10, 11, 12, 13, 14, 15);
    }

    const int rbase = tm * 16 + hi * 8;
    #pragma unroll
    for (int nt = 0; nt < NT; ++nt) {
        v8f acc = {};
        #pragma unroll
        for (int kt = 0; kt < KT; ++kt) {
            const v16h b = *(const v16h*)(Bp + ((size_t)((nt * KT + kt) * 32) + lane) * 16);
            acc = __builtin_amdgcn_wmma_f32_16x16x32_f16(
                false, afrag[kt], false, b, (short)0, acc, false, false);
        }
        const int col = nt * 16 + (lane & 15);
        const float bv = bias ? bias[col] : 0.0f;
        #pragma unroll
        for (int r = 0; r < 8; ++r) {
            float v = acc[r] + bv;
            if (act) v = (v > 0.0f) ? v : expm1f(v);
            const size_t idx = (size_t)(rbase + r) * Ncols + col;
            if (outF32) outF32[idx] = v;
            if (outF16) outF16[idx] = (_Float16)v;
        }
    }
}

// ---------------------------------------------------------------- edge phase
// One wave32 per edge. Lane owns channel pairs p = jj*32+lane (channels 2p,2p+1),
// head(p) = 2*jj + (lane>>4). Loads are b32 (v2h); reductions are 16-lane.
__global__ void edge_logits_kernel(const _Float16* __restrict__ xl,
                                   const _Float16* __restrict__ xr,
                                   const float*    __restrict__ att,    // [NH*NC] f32
                                   const int*      __restrict__ src,
                                   const int*      __restrict__ dst,
                                   float*          __restrict__ logits, // [EN*NH]
                                   float*          __restrict__ m,      // [N*NH]
                                   int E, int Nn) {
    const int lane = threadIdx.x & 31;
    const int e = blockIdx.x * blockDim.y + threadIdx.y;
    const int EN = E + Nn;
    if (e >= EN) return;
    int s, d;
    if (e < E) { s = src[e]; d = dst[e]; } else { s = e - E; d = s; }
    const v2h* __restrict__ ps = (const v2h*)(xl + (size_t)s * HC);
    const v2h* __restrict__ pd = (const v2h*)(xr + (size_t)d * HC);
    __builtin_prefetch(ps, 0, 0);            // global_prefetch_b8
    __builtin_prefetch(pd, 0, 0);

    float acc[4];
    #pragma unroll
    for (int jj = 0; jj < 4; ++jj) {
        const int p = jj * 32 + lane;
        const v2h a = ps[p];
        const v2h b = pd[p];
        float v0 = (float)a[0] + (float)b[0];
        float v1 = (float)a[1] + (float)b[1];
        v0 = (v0 > 0.0f) ? v0 : SLOPE * v0;
        v1 = (v1 > 0.0f) ? v1 : SLOPE * v1;
        acc[jj] = v0 * att[2 * p] + v1 * att[2 * p + 1];
    }
    #pragma unroll
    for (int jj = 0; jj < 4; ++jj) {
        #pragma unroll
        for (int off = 1; off < 16; off <<= 1)
            acc[jj] += __shfl_xor(acc[jj], off, 32);   // 16-lane groups
    }
    if ((lane & 15) == 0) {                  // lane 0 -> even heads, lane 16 -> odd heads
        const int hbase = (lane >> 4) & 1;
        #pragma unroll
        for (int jj = 0; jj < 4; ++jj) {
            const int h = 2 * jj + hbase;
            logits[(size_t)e * NH + h] = acc[jj];
            atomicMaxFloat(&m[(size_t)d * NH + h], acc[jj]);
        }
    }
}

// one thread per (edge, head): ex = exp(logit - m[d]); accumulate denom.
__global__ void edge_expsum_kernel(float* __restrict__ logits,
                                   const float* __restrict__ m,
                                   float* __restrict__ denom,
                                   const int* __restrict__ dst,
                                   int E, int Nn) {
    const int t = blockIdx.x * blockDim.x + threadIdx.x;
    const int EN = E + Nn;
    if (t >= EN * NH) return;
    const int e = t / NH, j = t - e * NH;
    const int d = (e < E) ? dst[e] : (e - E);
    const float ex = expf(logits[t] - m[(size_t)d * NH + j]);
    logits[t] = ex;
    atomicAdd(&denom[(size_t)d * NH + j], ex);
}

// one wave32 per edge: agg[d] += alpha * xl[s]  (global_atomic_add_f32)
__global__ void edge_aggregate_kernel(const _Float16* __restrict__ xl,
                                      const float* __restrict__ exlog,
                                      const float* __restrict__ denom,
                                      const int* __restrict__ src,
                                      const int* __restrict__ dst,
                                      float* __restrict__ agg,
                                      int E, int Nn) {
    const int lane = threadIdx.x & 31;
    const int e = blockIdx.x * blockDim.y + threadIdx.y;
    const int EN = E + Nn;
    if (e >= EN) return;
    int s, d;
    if (e < E) { s = src[e]; d = dst[e]; } else { s = e - E; d = s; }
    const v2h* __restrict__ ps = (const v2h*)(xl + (size_t)s * HC);
    __builtin_prefetch(ps, 0, 0);
    float* __restrict__ pd = agg + (size_t)d * HC;
    #pragma unroll
    for (int jj = 0; jj < 4; ++jj) {
        const int p = jj * 32 + lane;
        const int h = 2 * jj + ((lane >> 4) & 1);
        const float alpha = exlog[(size_t)e * NH + h] / denom[(size_t)d * NH + h];
        const v2h a = ps[p];
        atomicAdd(&pd[2 * p],     alpha * (float)a[0]);
        atomicAdd(&pd[2 * p + 1], alpha * (float)a[1]);
    }
}

// h = ELU(agg + bias_c + bias_o)  -> f16 for the MLP GEMMs
__global__ void combine_h_kernel(const float* __restrict__ agg,
                                 const float* __restrict__ bias_c,
                                 const float* __restrict__ bias_o,
                                 _Float16* __restrict__ h16, int n) {
    const int i = blockIdx.x * blockDim.x + threadIdx.x;
    if (i < n) {
        const int col = i & (HC - 1);
        float v = agg[i] + bias_c[col] + bias_o[col];
        v = (v > 0.0f) ? v : expm1f(v);
        h16[i] = (_Float16)v;
    }
}

// ---------------------------------------------------------------- launcher
extern "C" void kernel_launch(void* const* d_in, const int* in_sizes, int n_in,
                              void* d_out, int out_size, void* d_ws, size_t ws_size,
                              hipStream_t stream) {
    (void)in_sizes; (void)n_in; (void)out_size; (void)ws_size;

    const float* x       = (const float*)d_in[0];
    const int*   ei_c    = (const int*)d_in[1];   // [2,E] flat: src then dst
    const int*   ei_o    = (const int*)d_in[2];
    const float* Wl[2]   = { (const float*)d_in[3],  (const float*)d_in[9]  };
    const float* bl[2]   = { (const float*)d_in[4],  (const float*)d_in[10] };
    const float* Wr[2]   = { (const float*)d_in[5],  (const float*)d_in[11] };
    const float* br[2]   = { (const float*)d_in[6],  (const float*)d_in[12] };
    const float* att[2]  = { (const float*)d_in[7],  (const float*)d_in[13] };
    const float* biasC   = (const float*)d_in[8];
    const float* biasO   = (const float*)d_in[14];
    const float* W1      = (const float*)d_in[15];
    const float* b1      = (const float*)d_in[16];
    const float* W2      = (const float*)d_in[17];
    const float* b2      = (const float*)d_in[18];
    const int* srcs[2]   = { ei_c, ei_o };
    const int* dsts[2]   = { ei_c + NE, ei_o + NE };

    // ---- workspace carve (256B aligned)
    char*  ws  = (char*)d_ws;
    size_t off = 0;
    auto carve = [&](size_t bytes) -> void* {
        off = (off + 255) & ~(size_t)255;
        void* p = ws + off;
        off += bytes;
        return p;
    };
    _Float16* xf16   = (_Float16*)carve((size_t)N_NODES * DIN * 2);
    _Float16* wlp    = (_Float16*)carve((size_t)DIN * HC * 2);      // packed fragments
    _Float16* wrp    = (_Float16*)carve((size_t)DIN * HC * 2);
    _Float16* xl16   = (_Float16*)carve((size_t)N_NODES * HC * 2);
    _Float16* xr16   = (_Float16*)carve((size_t)N_NODES * HC * 2);
    float*    agg    = (float*)   carve((size_t)N_NODES * HC * 4);
    float*    logits = (float*)   carve((size_t)(NE + N_NODES) * NH * 4);
    float*    mbuf   = (float*)   carve((size_t)N_NODES * NH * 4);
    float*    denom  = (float*)   carve((size_t)N_NODES * NH * 4);
    _Float16* h16    = (_Float16*)carve((size_t)N_NODES * HC * 2);
    _Float16* w1p    = (_Float16*)carve((size_t)HC * NC * 2);
    _Float16* w2p    = (_Float16*)carve((size_t)NC * OUT_CH * 2);
    _Float16* h1f16  = (_Float16*)carve((size_t)N_NODES * NC * 2);

    const int  EN      = NE + N_NODES;
    const dim3 wblk(32, 8);                       // 8 wave32 per workgroup
    const int  eblocks = (EN + 7) / 8;
    const int  Mtiles  = N_NODES / 16;            // 3125 exactly
    const int  gblocks = (Mtiles + 7) / 8;

    // x -> f16 (once)
    { int n = N_NODES * DIN;
      cvt_f32_to_f16<<<(n + 255) / 256, 256, 0, stream>>>(x, xf16, n); }

    // zero shared aggregation buffer (both relations accumulate into it)
    { int n = N_NODES * HC;
      fill_f32<<<(n + 255) / 256, 256, 0, stream>>>(agg, 0.0f, n); }

    for (int r = 0; r < 2; ++r) {
        // pack weights into fragment order (folds f32->f16)
        { int n = DIN * HC;
          pack_B_kernel<<<(n + 255) / 256, 256, 0, stream>>>(Wl[r], wlp, DIN, HC);
          pack_B_kernel<<<(n + 255) / 256, 256, 0, stream>>>(Wr[r], wrp, DIN, HC); }

        // xl = x@Wl + bl ; xr = x@Wr + br   (WMMA, f16 outputs)
        wmma_gemm_f16<DIN / 32, HC / 16><<<gblocks, wblk, 0, stream>>>(
            xf16, wlp, bl[r], nullptr, xl16, Mtiles, 0);
        wmma_gemm_f16<DIN / 32, HC / 16><<<gblocks, wblk, 0, stream>>>(
            xf16, wrp, br[r], nullptr, xr16, Mtiles, 0);

        // segment-softmax state init
        { int n = N_NODES * NH;
          fill_f32<<<(n + 255) / 256, 256, 0, stream>>>(mbuf, -INFINITY, n);
          fill_f32<<<(n + 255) / 256, 256, 0, stream>>>(denom, 0.0f, n); }

        // attention logits + running max over incoming edges
        edge_logits_kernel<<<eblocks, wblk, 0, stream>>>(xl16, xr16, att[r],
                                                         srcs[r], dsts[r],
                                                         logits, mbuf, NE, N_NODES);
        // exp(logit - max), denom accumulation
        { int n = EN * NH;
          edge_expsum_kernel<<<(n + 255) / 256, 256, 0, stream>>>(logits, mbuf, denom,
                                                                  dsts[r], NE, N_NODES); }
        // agg[d] += alpha * xl[s]
        edge_aggregate_kernel<<<eblocks, wblk, 0, stream>>>(xl16, logits, denom,
                                                            srcs[r], dsts[r],
                                                            agg, NE, N_NODES);
    }

    // h = ELU(agg + bias_c + bias_o) -> f16
    { int n = N_NODES * HC;
      combine_h_kernel<<<(n + 255) / 256, 256, 0, stream>>>(agg, biasC, biasO, h16, n); }

    // MLP: h1 = ELU(h@W1 + b1) ; out = h1@W2 + b2
    { int n = HC * NC;
      pack_B_kernel<<<(n + 255) / 256, 256, 0, stream>>>(W1, w1p, HC, NC); }
    { int n = NC * OUT_CH;
      pack_B_kernel<<<(n + 255) / 256, 256, 0, stream>>>(W2, w2p, NC, OUT_CH); }

    wmma_gemm_f16<HC / 32, NC / 16><<<gblocks, wblk, 0, stream>>>(
        h16, w1p, b1, nullptr, h1f16, Mtiles, 1);
    wmma_gemm_f16<NC / 32, OUT_CH / 16><<<gblocks, wblk, 0, stream>>>(
        h1f16, w2p, b2, (float*)d_out, nullptr, Mtiles, 0);
}
